// MambaKoopmanWorldModel_86225763434830
// MI455X (gfx1250) — matrix operand order repaired and checked by reading, not compiled, not run on hardware
//
#include <hip/hip_runtime.h>
#include <hip/hip_bf16.h>
#include <math.h>

// ---------------------------------------------------------------------------
// Types for CDNA5 WMMA
// ---------------------------------------------------------------------------
typedef _Float16 v8h  __attribute__((ext_vector_type(8)));
typedef _Float16 v16h __attribute__((ext_vector_type(16)));
typedef float    v8f  __attribute__((ext_vector_type(8)));

#define EPI_NONE 0
#define EPI_GELU 1
#define EPI_SIG  2
#define EPI_TANH 3

__device__ __forceinline__ float apply_act(float v, int act) {
  switch (act) {
    case EPI_GELU: return 0.5f * v * (1.0f + erff(v * 0.70710678118654752f));
    case EPI_SIG:  return 1.0f / (1.0f + __expf(-v));
    case EPI_TANH: return tanhf(v);
    default:       return v;
  }
}

// ---------------------------------------------------------------------------
// Generic WMMA GEMM:  Out(M,N) = act( A(M,K) @ Wt(N,K)^T + bias + addb )
//   A   : f16 row-major (lda = K), with optional "skip last timestep" row remap
//   Wt  : f16, transposed weight, N x K row-major (K contiguous per output col)
//   out32 (f32) and/or out16 (f16), both M x N row-major, compact rows
// Block tile 128x128, K-step 32, 256 threads = 8 wave32 (4 M-waves x 2 N-waves)
// Each wave: 2x4 grid of 16x16 tiles -> eight v_wmma_f32_16x16x32_f16 per step.
// Double-buffered LDS + register-staged global loads: tile k+1 is in flight
// (global->VGPR) while the WMMAs for tile k execute; one barrier per K step.
// Tail handling: row pointers are clamped (garbage rows never stored), and the
// K-tail test is wave-uniform so the staged loads carry no exec-mask overhead.
// ---------------------------------------------------------------------------
__global__ __launch_bounds__(256, 2)
void gemm_wmma_f16(const _Float16* __restrict__ A,
                   const _Float16* __restrict__ Wt,
                   const float* __restrict__ bias,
                   const float* __restrict__ addb,
                   float* __restrict__ out32,
                   _Float16* __restrict__ out16,
                   int M, int N, int K, int remapT, int physT, int act)
{
  __shared__ alignas(64) _Float16 sA[2 * 128 * 32];
  __shared__ alignas(64) _Float16 sB[2 * 128 * 32];

  const int tid  = threadIdx.x;
  const int lane = tid & 31;
  const int wave = tid >> 5;     // 0..7
  const int wm   = wave >> 1;    // 0..3 : M offset = wm*32
  const int wn   = wave & 1;     // 0..1 : N offset = wn*64
  const int lm   = lane & 15;
  const int sel  = lane >> 4;    // 0/1 (lane half)

  const long blockM = (long)blockIdx.y * 128;
  const int  blockN = blockIdx.x * 128;

  v8f acc[2][4];
#pragma unroll
  for (int mi = 0; mi < 2; ++mi)
#pragma unroll
    for (int ni = 0; ni < 4; ++ni) {
      v8f z = {};
      acc[mi][ni] = z;
    }

  // ---- hoisted per-thread load descriptors (K-invariant: no div in loop) ----
  // Rows beyond M / cols beyond N are clamped to row 0: always-in-bounds loads
  // whose garbage results are discarded by the guarded epilogue stores.
  const int ldr = tid >> 2;          // 0..63
  const int c8  = (tid & 3) * 8;     // 0,8,16,24
  const _Float16* aRow[2];
  const _Float16* bRow[2];
#pragma unroll
  for (int h = 0; h < 2; ++h) {
    long gr = blockM + h * 64 + ldr;
    long pr = 0;
    if (gr < M) {
      pr = gr;
      if (remapT > 0) {              // logical row -> physical (skip last t)
        long bb2 = gr / remapT;
        pr = bb2 * (long)physT + (gr - bb2 * (long)remapT);
      }
    }
    aRow[h] = A + pr * (long)K;
    int gn = blockN + h * 64 + ldr;
    bRow[h] = Wt + (long)(gn < N ? gn : 0) * K;
  }

  // `full` is wave-uniform (k0, K scalar): no per-lane predication on the
  // fast path.  Scalar zero-fill path only runs for the K=16 tail.
  auto loadA = [&](int h, int k0, bool full) -> v8h {
    v8h av = {};
    int kb = k0 + c8;
    if (full) {
      av = *(const v8h*)(aRow[h] + kb);
    } else {
#pragma unroll
      for (int i = 0; i < 8; ++i) if (kb + i < K) av[i] = aRow[h][kb + i];
    }
    return av;
  };
  auto loadB = [&](int h, int k0, bool full) -> v8h {
    v8h bv = {};
    int kb = k0 + c8;
    if (full) {
      bv = *(const v8h*)(bRow[h] + kb);
    } else {
#pragma unroll
      for (int i = 0; i < 8; ++i) if (kb + i < K) bv[i] = bRow[h][kb + i];
    }
    return bv;
  };

  // ---- prologue: stage tile 0 in registers ----
  v8h ra[2], rb[2];
  {
    bool full0 = (32 <= K);
    ra[0] = loadA(0, 0, full0); ra[1] = loadA(1, 0, full0);
    rb[0] = loadB(0, 0, full0); rb[1] = loadB(1, 0, full0);
  }

  int p = 0;
  for (int k0 = 0; k0 < K; k0 += 32) {
    _Float16* pA = sA + p * (128 * 32);
    _Float16* pB = sB + p * (128 * 32);
#pragma unroll
    for (int h = 0; h < 2; ++h) {
      *(v8h*)(&pA[(h * 64 + ldr) * 32 + c8]) = ra[h];
      *(v8h*)(&pB[(h * 64 + ldr) * 32 + c8]) = rb[h];
    }
    // issue next tile's global loads; they fly across the barrier + WMMAs
    int k1 = k0 + 32;
    if (k1 < K) {
      bool full1 = (k1 + 32 <= K);
      ra[0] = loadA(0, k1, full1); ra[1] = loadA(1, k1, full1);
      rb[0] = loadB(0, k1, full1); rb[1] = loadB(1, k1, full1);
      __builtin_prefetch(aRow[0] + k1 + c8 + 32, 0, 0);
    }
    __syncthreads();   // compiler emits s_wait_dscnt 0 before signal: buffer-safe

    // ---- fragments: contiguous k order (k = 16*sel + s), same for A and B ----
    // A 16x32: lane<16 -> K 0..15; lane>=16 -> K 16..31 (contiguous)
    v16h af[2];
#pragma unroll
    for (int mi = 0; mi < 2; ++mi) {
      int r = wm * 32 + mi * 16 + lm;
      v8h lo = *(const v8h*)(&pA[r * 32 + sel * 16]);
      v8h hi = *(const v8h*)(&pA[r * 32 + sel * 16 + 8]);
      af[mi] = __builtin_shufflevector(lo, hi,
               0,1,2,3,4,5,6,7,8,9,10,11,12,13,14,15);
    }
    // B 32x16: lane = N%16, lane-half selects K 0..15 / 16..31 (contiguous)
    v16h bf[4];
#pragma unroll
    for (int ni = 0; ni < 4; ++ni) {
      int n = wn * 64 + ni * 16 + lm;
      bf[ni] = *(const v16h*)(&pB[n * 32 + sel * 16]);
    }
#pragma unroll
    for (int mi = 0; mi < 2; ++mi)
#pragma unroll
      for (int ni = 0; ni < 4; ++ni)
        acc[mi][ni] = __builtin_amdgcn_wmma_f32_16x16x32_f16(
            false, af[mi], false, bf[ni], (short)0, acc[mi][ni], false, false);
    p ^= 1;
  }

  // ---- epilogue: C/D layout VGPR r -> M = r + 8*sel, N = lane%16 ----
#pragma unroll
  for (int mi = 0; mi < 2; ++mi) {
#pragma unroll
    for (int ni = 0; ni < 4; ++ni) {
      int col = blockN + wn * 64 + ni * 16 + lm;
      if (col >= N) continue;
      float bv = bias ? bias[col] : 0.0f;
      v8f a = acc[mi][ni];
#pragma unroll
      for (int r = 0; r < 8; ++r) {
        long row = blockM + wm * 32 + mi * 16 + sel * 8 + r;
        if (row >= M) continue;
        float v = a[r] + bv;
        if (addb)  v += addb[row * (long)N + col];
        v = apply_act(v, act);
        if (out32) out32[row * (long)N + col] = v;
        if (out16) out16[row * (long)N + col] = (_Float16)v;
      }
    }
  }
}

// ---------------------------------------------------------------------------
// Weight convert: f32 (K x N) -> f16 transposed (N x K)
// ---------------------------------------------------------------------------
__global__ void k_transpose_w(const float* __restrict__ in,
                              _Float16* __restrict__ out, int K, int N) {
  int tid = blockIdx.x * blockDim.x + threadIdx.x;
  if (tid >= K * N) return;
  int n = tid / K, k = tid - n * K;
  out[tid] = (_Float16)in[(long)k * N + n];
}

__global__ void k_conv_f16(const float* __restrict__ in,
                           _Float16* __restrict__ out, long n) {
  long tid = blockIdx.x * (long)blockDim.x + threadIdx.x;
  if (tid < n) out[tid] = (_Float16)in[tid];
}

// ---------------------------------------------------------------------------
// Selective scan: s_t = a*s + (1-a)*b, fused with cs = c * s  (f16 out)
// One thread per (batch, state) element; coalesced across S each step.
// ---------------------------------------------------------------------------
__global__ void k_scan(const float* __restrict__ a, const float* __restrict__ b,
                       const float* __restrict__ c, _Float16* __restrict__ cs,
                       int B, int T, int S) {
  int tid = blockIdx.x * blockDim.x + threadIdx.x;
  if (tid >= B * S) return;
  int bb = tid / S, ss = tid - bb * S;
  long base = (long)bb * T * S + ss;
  float s = 0.0f;
  for (int t = 0; t < T; ++t) {
    long i = base + (long)t * S;
    float av = a[i];
    s = av * s + (1.0f - av) * b[i];
    cs[i] = (_Float16)(c[i] * s);
  }
}

// ---------------------------------------------------------------------------
// Driver
// ---------------------------------------------------------------------------
extern "C" void kernel_launch(void* const* d_in, const int* in_sizes, int n_in,
                              void* d_out, int out_size, void* d_ws, size_t ws_size,
                              hipStream_t stream) {
  constexpr int kB = 16, kT = 2048, kObs = 64, kAct = 16, kD = 1024, kS = 256, kL = 256;
  constexpr long kBT  = (long)kB * kT;        // 32768
  constexpr long kBT1 = (long)kB * (kT - 1);  // 32752

  const float* o_seq = (const float*)d_in[0];
  const float* u_seq = (const float*)d_in[1];
  const float* oe_W1 = (const float*)d_in[2];  const float* oe_b1 = (const float*)d_in[3];
  const float* oe_W2 = (const float*)d_in[4];  const float* oe_b2 = (const float*)d_in[5];
  const float* ae_W1 = (const float*)d_in[6];  const float* ae_b1 = (const float*)d_in[7];
  const float* ae_W2 = (const float*)d_in[8];  const float* ae_b2 = (const float*)d_in[9];
  const float* Wa = (const float*)d_in[10];    const float* ba = (const float*)d_in[11];
  const float* Wb = (const float*)d_in[12];    const float* bb = (const float*)d_in[13];
  const float* Wc = (const float*)d_in[14];    const float* bc = (const float*)d_in[15];
  const float* Wd = (const float*)d_in[16];    const float* bd = (const float*)d_in[17];
  const float* Wy = (const float*)d_in[18];    const float* by = (const float*)d_in[19];
  const float* Wl = (const float*)d_in[20];    const float* bl = (const float*)d_in[21];
  const float* Am = (const float*)d_in[22];
  const float* Bm = (const float*)d_in[23];
  const float* dec_W1 = (const float*)d_in[24]; const float* dec_b1 = (const float*)d_in[25];
  const float* dec_W2 = (const float*)d_in[26]; const float* dec_b2 = (const float*)d_in[27];

  float* h_out    = (float*)d_out;                       // (B,T,L)
  float* hlin_out = h_out + kBT * kL;                    // (B,T-1,L)
  float* ohat_out = hlin_out + kBT1 * kL;                // (B,T-1,OBS)

  // ---- bump allocator on d_ws ----
  char* ws = (char*)d_ws;
  size_t off = 0;
  auto alloc = [&](size_t bytes) -> void* {
    off = (off + 255) & ~(size_t)255;
    void* p = ws + off;
    off += bytes;
    return p;
  };

  // ---- convert weights: f32 (K,N) -> f16 (N,K) transposed ----
  auto tr = [&](const float* w, int Kd, int Nd) -> _Float16* {
    _Float16* o = (_Float16*)alloc((size_t)Kd * Nd * sizeof(_Float16));
    int n = Kd * Nd;
    k_transpose_w<<<(n + 255) / 256, 256, 0, stream>>>(w, o, Kd, Nd);
    return o;
  };
  _Float16* oeW1t = tr(oe_W1, kObs, kD);
  _Float16* oeW2t = tr(oe_W2, kD, kD);
  _Float16* aeW1t = tr(ae_W1, kAct, kD);
  _Float16* aeW2t = tr(ae_W2, kD, kD);
  _Float16* Wat   = tr(Wa, kD, kS);
  _Float16* Wbt   = tr(Wb, kD, kS);
  _Float16* Wct   = tr(Wc, kD, kS);
  _Float16* Wdt   = tr(Wd, kD, kD);
  _Float16* Wyt   = tr(Wy, kS, kD);
  _Float16* Wlt   = tr(Wl, kD, kL);
  _Float16* At    = tr(Am, kL, kL);
  _Float16* Bmt   = tr(Bm, kAct, kL);
  _Float16* dW1t  = tr(dec_W1, kL, kD);
  _Float16* dW2t  = tr(dec_W2, kD, kObs);

  // ---- activation buffers ----
  _Float16* o16   = (_Float16*)alloc(kBT * kObs * sizeof(_Float16));
  _Float16* u16   = (_Float16*)alloc(kBT * kAct * sizeof(_Float16));
  _Float16* t16   = (_Float16*)alloc(kBT * kD * sizeof(_Float16));   // hidden MLP tmp
  float*    buf32 = (float*)   alloc(kBT * kD * sizeof(float));      // x32 / d32 / hl32
  _Float16* x16   = (_Float16*)alloc(kBT * kD * sizeof(_Float16));   // x, then y
  float*    a32   = (float*)   alloc(kBT * kS * sizeof(float));
  float*    b32   = (float*)   alloc(kBT * kS * sizeof(float));
  float*    c32   = (float*)   alloc(kBT * kS * sizeof(float));
  _Float16* cs16  = (_Float16*)alloc(kBT * kS * sizeof(_Float16));
  _Float16* h16   = (_Float16*)alloc(kBT * kL * sizeof(_Float16));
  _Float16* hl16  = (_Float16*)alloc(kBT1 * kL * sizeof(_Float16));
  (void)ws_size; (void)in_sizes; (void)n_in; (void)out_size;

  {
    long n = kBT * kObs;
    k_conv_f16<<<(int)((n + 255) / 256), 256, 0, stream>>>(o_seq, o16, n);
    n = kBT * kAct;
    k_conv_f16<<<(int)((n + 255) / 256), 256, 0, stream>>>(u_seq, u16, n);
  }

  auto gemm = [&](const _Float16* Ap, const _Float16* Wtp, const float* bias,
                  const float* addbp, float* o32, _Float16* o16p,
                  long M, int N, int K, int remapT, int physT, int act) {
    dim3 g((N + 127) / 128, (unsigned)((M + 127) / 128));
    gemm_wmma_f16<<<g, 256, 0, stream>>>(Ap, Wtp, bias, addbp, o32, o16p,
                                         (int)M, N, K, remapT, physT, act);
  };

  // encoders: x = (gelu(o@W1+b1)@W2+b2) + (gelu(u@W1'+b1')@W2'+b2')
  gemm(o16, oeW1t, oe_b1, nullptr, nullptr, t16, kBT, kD, kObs, 0, 0, EPI_GELU);
  gemm(t16, oeW2t, oe_b2, nullptr, buf32, nullptr, kBT, kD, kD, 0, 0, EPI_NONE);
  gemm(u16, aeW1t, ae_b1, nullptr, nullptr, t16, kBT, kD, kAct, 0, 0, EPI_GELU);
  gemm(t16, aeW2t, ae_b2, buf32, nullptr, x16, kBT, kD, kD, 0, 0, EPI_NONE);

  // selection projections
  gemm(x16, Wat, ba, nullptr, a32, nullptr, kBT, kS, kD, 0, 0, EPI_SIG);
  gemm(x16, Wbt, bb, nullptr, b32, nullptr, kBT, kS, kD, 0, 0, EPI_NONE);
  gemm(x16, Wct, bc, nullptr, c32, nullptr, kBT, kS, kD, 0, 0, EPI_TANH);
  gemm(x16, Wdt, bd, nullptr, buf32, nullptr, kBT, kD, kD, 0, 0, EPI_NONE);  // d_skip

  // selective scan (fused c*s -> f16)
  k_scan<<<(kB * kS + 255) / 256, 256, 0, stream>>>(a32, b32, c32, cs16, kB, kT, kS);

  // y = (c*s)@Wy + by + d_skip  (reuse x16 as y16)
  gemm(cs16, Wyt, by, buf32, nullptr, x16, kBT, kD, kS, 0, 0, EPI_NONE);

  // h = y@Wl + bl  -> output 0 (f32) + f16 copy for chaining
  gemm(x16, Wlt, bl, nullptr, h_out, h16, kBT, kL, kD, 0, 0, EPI_NONE);

  // h_lin = h[:, :-1]@A + u[:, :-1]@Bm  (row remap T-1 over physical T)
  gemm(h16, At, nullptr, nullptr, buf32, nullptr, kBT1, kL, kL, kT - 1, kT, EPI_NONE);
  gemm(u16, Bmt, nullptr, buf32, hlin_out, hl16, kBT1, kL, kAct, kT - 1, kT, EPI_NONE);

  // decoder
  gemm(hl16, dW1t, dec_b1, nullptr, nullptr, t16, kBT1, kD, kL, 0, 0, EPI_GELU);
  gemm(t16, dW2t, dec_b2, nullptr, ohat_out, nullptr, kBT1, kObs, kD, 0, 0, EPI_NONE);
}
